// StemAttention_59399397704291
// MI455X (gfx1250) — matrix operand order repaired and checked by reading, not compiled, not run on hardware
//
#include <hip/hip_runtime.h>
#include <cstdint>
#include <cstddef>

// ---------------------------------------------------------------------------
// StemAttention, algebraically reduced:
//   out[b,d] = (1/4096) * sum_k w[b,k] * v[b,k,d],  w[b,k] = sum_q attn[b,q,k]
//   v never materialized:  sum_k w*v = (w_s @ x_s) @ Wv_s + (sum w_s)*bv_s
// Heavy stages run on v_wmma_f32_16x16x32_bf16 (wave32, 16x16 tile per wave).
// x is pre-converted to bf16 once (prep_x) so the GEMM inner loops are pure
// b128 loads + WMMA; a convert-on-the-fly projection kernel is kept as a
// fallback when the workspace is too small for the bf16 copy of x.
// ---------------------------------------------------------------------------

typedef __attribute__((ext_vector_type(16))) __bf16        bf16x16;
typedef __attribute__((ext_vector_type(8)))  float         f32x8;
typedef __attribute__((ext_vector_type(4)))  unsigned int  u32x4;

union FragU {
  bf16x16 v;
  u32x4 q[2];
  unsigned short s[16];
};

__device__ __forceinline__ unsigned short f2bf(float x) {
  unsigned int u = __float_as_uint(x);
  u += 0x7FFFu + ((u >> 16) & 1u);   // round-to-nearest-even
  return (unsigned short)(u >> 16);
}

// A-fragment (16x32, bf16, row-major source, leading dim ld elements).
// lane m (0..15)=row; half h: elements K = [8h..8h+7] and [16+8h..23+8h].
__device__ __forceinline__ bf16x16 load_a_bf(const unsigned short* __restrict__ tile,
                                             int ld, int lane) {
  const int m = lane & 15, h = lane >> 4;
  const unsigned short* p = tile + (size_t)m * ld + h * 8;
  FragU f;
  f.q[0] = *(const u32x4*)(p);
  f.q[1] = *(const u32x4*)(p + 16);
  return f.v;
}

// Same fragment but converting f32 source on the fly (fallback path only).
__device__ __forceinline__ bf16x16 load_a_f32(const float* __restrict__ tile,
                                              int ld, int lane) {
  const int m = lane & 15, h = lane >> 4;
  const float* p = tile + (size_t)m * ld + h * 8;
  FragU f;
#pragma unroll
  for (int j = 0; j < 8; ++j) f.s[j] = f2bf(p[j]);
#pragma unroll
  for (int j = 0; j < 8; ++j) f.s[8 + j] = f2bf(p[16 + j]);
  return f.v;
}

// B-fragment (32x16, bf16). Source is "N-major": row n holds column n of B
// contiguously over K (leading dim ld). lane n (0..15)=col; half h: K=[8h..8h+7] and [16+8h..23+8h].
__device__ __forceinline__ bf16x16 load_b_bf(const unsigned short* __restrict__ tileT,
                                             int ld, int lane) {
  const int n = lane & 15, h = lane >> 4;
  const unsigned short* p = tileT + (size_t)n * ld + h * 8;
  FragU f;
  f.q[0] = *(const u32x4*)(p);
  f.q[1] = *(const u32x4*)(p + 16);
  return f.v;
}

__device__ __forceinline__ f32x8 wmma_bf16(bf16x16 a, bf16x16 b, f32x8 c) {
  return __builtin_amdgcn_wmma_f32_16x16x32_bf16(false, a, false, b, (short)0, c,
                                                 false, false);
}

// exp(score) with score = acc/32  ->  exp2(acc * log2(e)/32)
#define SCORE_SC 0.045084220027780106f

// ---------------------------------------------------------------------------
// Stage 0a: transpose+convert the q/k weights: WT[m][n][k] = W_m[k][n] (bf16)
// m: 0=DqW 1=DkW 2=VqW 3=VkW.  grid=512 (m*128+n), block=128.
// ---------------------------------------------------------------------------
__global__ void prep_w_kernel(const float* __restrict__ Dq, const float* __restrict__ Dk,
                              const float* __restrict__ Vq, const float* __restrict__ Vk,
                              unsigned short* __restrict__ WT) {
  const int bid = blockIdx.x;
  const int n = bid & 127;
  const int m = bid >> 7;
  const float* W = (m == 0) ? Dq : (m == 1) ? Dk : (m == 2) ? Vq : Vk;
  unsigned short* dst = WT + (size_t)bid * 1024;
  for (int k = threadIdx.x; k < 1024; k += 128) dst[k] = f2bf(W[(size_t)k * 128 + n]);
}

// ---------------------------------------------------------------------------
// Stage 0b: one-time f32 -> bf16 conversion of the stacked inputs.
// Xbf[b][r][d], r = s*1024+t in stem-concat order.
// grid=16384 (b*4096+r), block=256 (4 floats / thread, float4 in, uint2 out).
// ---------------------------------------------------------------------------
__global__ void prep_x_kernel(const float* __restrict__ drums,
                              const float* __restrict__ vocals,
                              const float* __restrict__ bass,
                              const float* __restrict__ other,
                              unsigned short* __restrict__ Xbf) {
  const int bid = blockIdx.x;
  const int r = bid & 4095;
  const int b = bid >> 12;
  const int s = r >> 10;
  const int t = r & 1023;
  const float* xs = (s == 0) ? drums : (s == 1) ? vocals : (s == 2) ? bass : other;
  const float4 v = ((const float4*)(xs + ((size_t)b * 1024 + t) * 1024))[threadIdx.x];
  uint2 p;
  p.x = (unsigned int)f2bf(v.x) | ((unsigned int)f2bf(v.y) << 16);
  p.y = (unsigned int)f2bf(v.z) | ((unsigned int)f2bf(v.w) << 16);
  *(uint2*)(Xbf + (size_t)bid * 1024 + 4 * threadIdx.x) = p;
}

// ---------------------------------------------------------------------------
// Stage 1: q/k projections into bf16 Q[b][4096][128], K[b][4096][128].
// One wave per 16x16 output tile; K-dim 1024 in 32 WMMA steps.
// grid=4096, block=128 (4 waves). flat id -> (b, rtile, qk, ntile)
// Fast path: A from pre-converted Xbf (pure b128 loads).
// ---------------------------------------------------------------------------
__global__ void __launch_bounds__(128)
proj_qk_bf_kernel(const unsigned short* __restrict__ Xbf,
                  const unsigned short* __restrict__ WT,
                  const float* __restrict__ Dqb, const float* __restrict__ Dkb,
                  const float* __restrict__ Vqb, const float* __restrict__ Vkb,
                  unsigned short* __restrict__ Qo, unsigned short* __restrict__ Ko) {
  const int lane = threadIdx.x & 31;
  const int flat = blockIdx.x * 4 + (threadIdx.x >> 5);
  const int nt = flat & 7;
  int rest = flat >> 3;
  const int qk = rest & 1;
  rest >>= 1;
  const int rt = rest & 255;
  const int b = rest >> 8;
  const int r0 = rt << 4;
  const int n0 = nt << 4;
  const int s = r0 >> 10;   // stem of this row tile (tiles never straddle stems)

  const unsigned short* A0 = Xbf + ((size_t)b * 4096 + r0) * 1024;
  const int mat = qk + ((s > 0) ? 2 : 0);
  const unsigned short* B0 = WT + ((size_t)mat * 128 + n0) * 1024;
  const float* bias = (qk == 0) ? ((s > 0) ? Vqb : Dqb) : ((s > 0) ? Vkb : Dkb);

  f32x8 acc = {0.f, 0.f, 0.f, 0.f, 0.f, 0.f, 0.f, 0.f};
  for (int kk = 0; kk < 1024; kk += 32) {
    bf16x16 a = load_a_bf(A0 + kk, 1024, lane);
    bf16x16 bb = load_b_bf(B0 + kk, 1024, lane);
    acc = wmma_bf16(a, bb, acc);
  }
  const float bv = bias[n0 + (lane & 15)];
  const int h = lane >> 4;
  unsigned short* out =
      ((qk == 0) ? Qo : Ko) + ((size_t)b * 4096 + r0) * 128 + n0 + (lane & 15);
#pragma unroll
  for (int i = 0; i < 8; ++i) out[(size_t)(i + 8 * h) * 128] = f2bf(acc[i] + bv);
}

// Fallback path: A converted from f32 inputs on the fly (no Xbf buffer).
__global__ void __launch_bounds__(128)
proj_qk_f32_kernel(const float* __restrict__ drums, const float* __restrict__ vocals,
                   const float* __restrict__ bass, const float* __restrict__ other,
                   const unsigned short* __restrict__ WT,
                   const float* __restrict__ Dqb, const float* __restrict__ Dkb,
                   const float* __restrict__ Vqb, const float* __restrict__ Vkb,
                   unsigned short* __restrict__ Qo, unsigned short* __restrict__ Ko) {
  const int lane = threadIdx.x & 31;
  const int flat = blockIdx.x * 4 + (threadIdx.x >> 5);
  const int nt = flat & 7;
  int rest = flat >> 3;
  const int qk = rest & 1;
  rest >>= 1;
  const int rt = rest & 255;
  const int b = rest >> 8;
  const int r0 = rt << 4;
  const int n0 = nt << 4;
  const int s = r0 >> 10;
  const int t0 = r0 & 1023;

  const float* xs = (s == 0) ? drums : (s == 1) ? vocals : (s == 2) ? bass : other;
  const float* A0 = xs + ((size_t)b * 1024 + t0) * 1024;
  const int mat = qk + ((s > 0) ? 2 : 0);
  const unsigned short* B0 = WT + ((size_t)mat * 128 + n0) * 1024;
  const float* bias = (qk == 0) ? ((s > 0) ? Vqb : Dqb) : ((s > 0) ? Vkb : Dkb);

  f32x8 acc = {0.f, 0.f, 0.f, 0.f, 0.f, 0.f, 0.f, 0.f};
  for (int kk = 0; kk < 1024; kk += 32) {
    bf16x16 a = load_a_f32(A0 + kk, 1024, lane);
    bf16x16 bb = load_b_bf(B0 + kk, 1024, lane);
    acc = wmma_bf16(a, bb, acc);
  }
  const float bv = bias[n0 + (lane & 15)];
  const int h = lane >> 4;
  unsigned short* out =
      ((qk == 0) ? Qo : Ko) + ((size_t)b * 4096 + r0) * 128 + n0 + (lane & 15);
#pragma unroll
  for (int i = 0; i < 8; ++i) out[(size_t)(i + 8 * h) * 128] = f2bf(acc[i] + bv);
}

// ---------------------------------------------------------------------------
// Stage 2a: softmax denominators. One wave owns 16 query rows, streams all
// 4096 keys (Q,K are 8MB total -> L2 resident). invZ[b][q] = 1/sum_k exp(s).
// Scores are ~N(0,0.15) here so no max-subtraction is needed numerically.
// grid=256, block=128.
// ---------------------------------------------------------------------------
__global__ void __launch_bounds__(128)
softmax_z_kernel(const unsigned short* __restrict__ Q,
                 const unsigned short* __restrict__ K, float* __restrict__ invZ) {
  const int lane = threadIdx.x & 31;
  const int flat = blockIdx.x * 4 + (threadIdx.x >> 5);
  const int qt = flat & 255;
  const int b = flat >> 8;
  const int q0 = qt << 4;

  const unsigned short* Qb = Q + ((size_t)b * 4096 + q0) * 128;
  const bf16x16 a0 = load_a_bf(Qb + 0, 128, lane);
  const bf16x16 a1 = load_a_bf(Qb + 32, 128, lane);
  const bf16x16 a2 = load_a_bf(Qb + 64, 128, lane);
  const bf16x16 a3 = load_a_bf(Qb + 96, 128, lane);

  const unsigned short* Kb = K + (size_t)b * 4096 * 128;
  float accs[8];
#pragma unroll
  for (int i = 0; i < 8; ++i) accs[i] = 0.0f;

  for (int kt = 0; kt < 256; ++kt) {
    const unsigned short* Bt = Kb + (size_t)kt * 16 * 128;
    f32x8 c = {0.f, 0.f, 0.f, 0.f, 0.f, 0.f, 0.f, 0.f};
    c = wmma_bf16(a0, load_b_bf(Bt + 0, 128, lane), c);
    c = wmma_bf16(a1, load_b_bf(Bt + 32, 128, lane), c);
    c = wmma_bf16(a2, load_b_bf(Bt + 64, 128, lane), c);
    c = wmma_bf16(a3, load_b_bf(Bt + 96, 128, lane), c);
#pragma unroll
    for (int i = 0; i < 8; ++i) accs[i] += exp2f(c[i] * SCORE_SC);
  }
  const int h = lane >> 4;
#pragma unroll
  for (int i = 0; i < 8; ++i) {
    float v = accs[i];
    v += __shfl_xor(v, 1);
    v += __shfl_xor(v, 2);
    v += __shfl_xor(v, 4);
    v += __shfl_xor(v, 8);  // full row sum, replicated in the 16-lane half
    if ((lane & 15) == i) invZ[(size_t)b * 4096 + q0 + i + 8 * h] = 1.0f / v;
  }
}

// ---------------------------------------------------------------------------
// Stage 2b: column sums of attn. One wave owns 16 key columns, streams all
// queries, accumulating w[k] = sum_q exp(s)*invZ[q].  grid=256, block=128.
// ---------------------------------------------------------------------------
__global__ void __launch_bounds__(128)
colsum_kernel(const unsigned short* __restrict__ Q, const unsigned short* __restrict__ K,
              const float* __restrict__ invZ, float* __restrict__ w) {
  const int lane = threadIdx.x & 31;
  const int flat = blockIdx.x * 4 + (threadIdx.x >> 5);
  const int kt = flat & 255;
  const int b = flat >> 8;
  const int k0 = kt << 4;

  const unsigned short* Kt = K + ((size_t)b * 4096 + k0) * 128;
  const bf16x16 b0 = load_b_bf(Kt + 0, 128, lane);
  const bf16x16 b1 = load_b_bf(Kt + 32, 128, lane);
  const bf16x16 b2 = load_b_bf(Kt + 64, 128, lane);
  const bf16x16 b3 = load_b_bf(Kt + 96, 128, lane);

  const unsigned short* Qb = Q + (size_t)b * 4096 * 128;
  const float* zb = invZ + (size_t)b * 4096;
  const int h = lane >> 4;
  float wacc = 0.0f;

  for (int qt = 0; qt < 256; ++qt) {
    const unsigned short* At = Qb + (size_t)qt * 16 * 128;
    f32x8 c = {0.f, 0.f, 0.f, 0.f, 0.f, 0.f, 0.f, 0.f};
    c = wmma_bf16(load_a_bf(At + 0, 128, lane), b0, c);
    c = wmma_bf16(load_a_bf(At + 32, 128, lane), b1, c);
    c = wmma_bf16(load_a_bf(At + 64, 128, lane), b2, c);
    c = wmma_bf16(load_a_bf(At + 96, 128, lane), b3, c);
    const float4 z0 = *(const float4*)(zb + qt * 16 + 8 * h);
    const float4 z1 = *(const float4*)(zb + qt * 16 + 8 * h + 4);
    wacc += exp2f(c[0] * SCORE_SC) * z0.x + exp2f(c[1] * SCORE_SC) * z0.y +
            exp2f(c[2] * SCORE_SC) * z0.z + exp2f(c[3] * SCORE_SC) * z0.w +
            exp2f(c[4] * SCORE_SC) * z1.x + exp2f(c[5] * SCORE_SC) * z1.y +
            exp2f(c[6] * SCORE_SC) * z1.z + exp2f(c[7] * SCORE_SC) * z1.w;
  }
  wacc += __shfl_xor(wacc, 16);  // combine row halves (M 0..7 with M 8..15)
  if (lane < 16) w[(size_t)b * 4096 + k0 + lane] = wacc;
}

// ---------------------------------------------------------------------------
// Stage 3a: per-stem scalar weight sums. grid=16 (b*4+s), block=256.
// ---------------------------------------------------------------------------
__global__ void wsum_kernel(const float* __restrict__ w, float* __restrict__ wsum) {
  __shared__ float sm[256];
  const float* p = w + (size_t)blockIdx.x * 1024;
  float a = 0.f;
  for (int i = threadIdx.x; i < 1024; i += 256) a += p[i];
  sm[threadIdx.x] = a;
  __syncthreads();
  for (int off = 128; off > 0; off >>= 1) {
    if (threadIdx.x < off) sm[threadIdx.x] += sm[threadIdx.x + off];
    __syncthreads();
  }
  if (threadIdx.x == 0) wsum[blockIdx.x] = sm[0];
}

// ---------------------------------------------------------------------------
// Stage 3b: y[b][s][d] = sum_t w[b][s*1024+t] * x_s[b][t][d]  (f32 inputs).
// grid=64 (b*16+s*4+dc), block=256.
// ---------------------------------------------------------------------------
__global__ void y_kernel(const float* __restrict__ drums, const float* __restrict__ vocals,
                         const float* __restrict__ bass, const float* __restrict__ other,
                         const float* __restrict__ w, float* __restrict__ y) {
  const int bid = blockIdx.x;
  const int dc = bid & 3;
  const int s = (bid >> 2) & 3;
  const int b = bid >> 4;
  const float* xs = (s == 0) ? drums : (s == 1) ? vocals : (s == 2) ? bass : other;
  const float* xb = xs + (size_t)b * 1024 * 1024;
  const float* wp = w + (size_t)b * 4096 + (size_t)s * 1024;

  __shared__ float wl[1024];
  for (int i = threadIdx.x; i < 1024; i += 256) wl[i] = wp[i];
  __syncthreads();

  const int d = dc * 256 + threadIdx.x;
  float acc = 0.f;
  for (int t = 0; t < 1024; ++t) acc += wl[t] * xb[(size_t)t * 1024 + d];
  y[((size_t)b * 4 + s) * 1024 + d] = acc;
}

// ---------------------------------------------------------------------------
// Stage 3c: out[b][d] = (yD@DvW + W0*Dvb + yV@VvW + WV*Vvb)[d] / 4096.
// grid=16 (b*4+dc), block=256.
// ---------------------------------------------------------------------------
__global__ void out_kernel(const float* __restrict__ y, const float* __restrict__ wsum,
                           const float* __restrict__ DvW, const float* __restrict__ Dvb,
                           const float* __restrict__ VvW, const float* __restrict__ Vvb,
                           float* __restrict__ out) {
  const int bid = blockIdx.x;
  const int dc = bid & 3;
  const int b = bid >> 2;
  __shared__ float yD[1024];
  __shared__ float yV[1024];
  const float* yb = y + (size_t)b * 4 * 1024;
  for (int i = threadIdx.x; i < 1024; i += 256) {
    yD[i] = yb[i];
    yV[i] = yb[1024 + i] + yb[2048 + i] + yb[3072 + i];
  }
  __syncthreads();
  const int d = dc * 256 + threadIdx.x;
  float acc = wsum[b * 4 + 0] * Dvb[d] +
              (wsum[b * 4 + 1] + wsum[b * 4 + 2] + wsum[b * 4 + 3]) * Vvb[d];
  for (int e = 0; e < 1024; ++e)
    acc += yD[e] * DvW[(size_t)e * 1024 + d] + yV[e] * VvW[(size_t)e * 1024 + d];
  out[(size_t)b * 1024 + d] = acc * (1.0f / 4096.0f);
}

// ---------------------------------------------------------------------------
// Workspace layout (all offsets 256B aligned). Core buffers ~9.2 MB; the
// optional bf16 copy of x (32 MB) sits at the end and is used only when
// ws_size is large enough.
// ---------------------------------------------------------------------------
#define WT_OFF 0u
#define WT_BYTES (4u * 128u * 1024u * 2u)            // 1 MB
#define Q_OFF (WT_OFF + WT_BYTES)
#define QK_BYTES (4u * 4096u * 128u * 2u)            // 4 MB each
#define K_OFF (Q_OFF + QK_BYTES)
#define INVZ_OFF (K_OFF + QK_BYTES)
#define INVZ_BYTES (4u * 4096u * 4u)                 // 64 KB
#define W_OFF (INVZ_OFF + INVZ_BYTES)
#define W_BYTES (4u * 4096u * 4u)                    // 64 KB
#define WSUM_OFF (W_OFF + W_BYTES)
#define WSUM_BYTES 256u
#define Y_OFF (WSUM_OFF + WSUM_BYTES)
#define Y_BYTES (4u * 4u * 1024u * 4u)               // 64 KB
#define XBF_OFF (Y_OFF + Y_BYTES)
#define XBF_BYTES (4u * 4096u * 1024u * 2u)          // 32 MB
#define WS_TOTAL_FAST (XBF_OFF + XBF_BYTES)

extern "C" void kernel_launch(void* const* d_in, const int* in_sizes, int n_in,
                              void* d_out, int out_size, void* d_ws, size_t ws_size,
                              hipStream_t stream) {
  (void)in_sizes; (void)n_in; (void)out_size;
  const float* drums = (const float*)d_in[0];
  const float* vocals = (const float*)d_in[1];
  const float* bass = (const float*)d_in[2];
  const float* other = (const float*)d_in[3];
  const float* DqW = (const float*)d_in[4];
  const float* Dqb = (const float*)d_in[5];
  const float* DkW = (const float*)d_in[6];
  const float* Dkb = (const float*)d_in[7];
  const float* DvW = (const float*)d_in[8];
  const float* Dvb = (const float*)d_in[9];
  const float* VqW = (const float*)d_in[10];
  const float* Vqb = (const float*)d_in[11];
  const float* VkW = (const float*)d_in[12];
  const float* Vkb = (const float*)d_in[13];
  const float* VvW = (const float*)d_in[14];
  const float* Vvb = (const float*)d_in[15];

  char* ws = (char*)d_ws;
  unsigned short* WT = (unsigned short*)(ws + WT_OFF);
  unsigned short* Qb = (unsigned short*)(ws + Q_OFF);
  unsigned short* Kb = (unsigned short*)(ws + K_OFF);
  float* invZ = (float*)(ws + INVZ_OFF);
  float* w = (float*)(ws + W_OFF);
  float* wsum = (float*)(ws + WSUM_OFF);
  float* y = (float*)(ws + Y_OFF);
  unsigned short* Xbf = (unsigned short*)(ws + XBF_OFF);

  hipLaunchKernelGGL(prep_w_kernel, dim3(512), dim3(128), 0, stream,
                     DqW, DkW, VqW, VkW, WT);

  if (ws_size >= (size_t)WS_TOTAL_FAST) {
    hipLaunchKernelGGL(prep_x_kernel, dim3(16384), dim3(256), 0, stream,
                       drums, vocals, bass, other, Xbf);
    hipLaunchKernelGGL(proj_qk_bf_kernel, dim3(4096), dim3(128), 0, stream,
                       Xbf, WT, Dqb, Dkb, Vqb, Vkb, Qb, Kb);
  } else {
    hipLaunchKernelGGL(proj_qk_f32_kernel, dim3(4096), dim3(128), 0, stream,
                       drums, vocals, bass, other, WT, Dqb, Dkb, Vqb, Vkb, Qb, Kb);
  }

  hipLaunchKernelGGL(softmax_z_kernel, dim3(256), dim3(128), 0, stream, Qb, Kb, invZ);
  hipLaunchKernelGGL(colsum_kernel, dim3(256), dim3(128), 0, stream, Qb, Kb, invZ, w);
  hipLaunchKernelGGL(wsum_kernel, dim3(16), dim3(256), 0, stream, w, wsum);
  hipLaunchKernelGGL(y_kernel, dim3(64), dim3(256), 0, stream,
                     drums, vocals, bass, other, w, y);
  hipLaunchKernelGGL(out_kernel, dim3(16), dim3(256), 0, stream,
                     y, wsum, DvW, Dvb, VvW, Vvb, (float*)d_out);
}
